// KoLeoLoss_44573170598814
// MI455X (gfx1250) — matrix operand order repaired and checked by reading, not compiled, not run on hardware
//
#include <hip/hip_runtime.h>
#include <hip/hip_bf16.h>

// KoLeo loss, MI455X (gfx1250), wave32 + WMMA bf16 path, async-LDS staging.
//
// Pipeline:
//   k_normalize   : row L2 norm -> inv_norm[i], bf16 normalized copy (16 MB, L2-resident)
//   k_init_keys   : packed argmax keys = 0
//   k_argmax_gram : fused  X*X^T  via v_wmma_f32_16x16x32_bf16 + running row-argmax,
//                   B panels staged global->LDS with ASYNC loads (depth-1 pipeline),
//                   diag excluded, merged across blocks with u64 atomicMax
//   k_distance    : exact f32  ||x_i - x_nn + eps||2  from raw input * inv_norm
//   k_mean        : mean(-log(dist+eps)) -> d_out[0]
//
// Workspace requirement: ~16.2 MB (xb bf16 | invn f32 | keys u64 | rowloss f32)

typedef __attribute__((ext_vector_type(16))) __bf16 v16bf;
typedef __attribute__((ext_vector_type(8)))  float  v8f;
typedef int v4i __attribute__((vector_size(16)));   // matches builtin param type

#define N_ROWS 8192
#define DIM    1024
#define EPSF   1e-8f

// ---- kernel 2 tiling ----
#define ROWS_PER_BLOCK 64      // 4 row sub-strips of 16
#define COL_CHUNK      1024    // columns per block (grid.y = 8 chunks)
#define COLS_PER_GROUP 128     // 2 column halves of 64 (waves 0-3 / 4-7)
#define N_GROUPS       (COL_CHUNK / COLS_PER_GROUP)
#define KSTEP          32
#define A_STRIDE_B     2064    // 1024*2 + 16B pad  -> conflict-free A fragments
#define B_COL_STRIDE_B 80      // 32*2  + 16B pad  -> conflict-free B fragments
// LDS: 64*2064 + 2*128*80 = 152576 B  (< 320 KB WGP LDS)

struct U32x8 { uint4 lo; uint4 hi; };   // 32B carrier for v16bf bit_cast

// ---- gfx1250 async global->LDS staging (ASYNCcnt path), with safe fallback ----
#if __has_builtin(__builtin_amdgcn_global_load_async_to_lds_b128)
#define USE_ASYNC_LDS 1
#else
#define USE_ASYNC_LDS 0
#endif

typedef __attribute__((address_space(1))) v4i glb_v4i;   // printed as "int4 __device__ *"
typedef __attribute__((address_space(3))) v4i lds_v4i;   // "__shared__" space

__device__ __forceinline__ glb_v4i* to_glb(const void* p) {
    return (glb_v4i*)(unsigned long long)(uintptr_t)p;
}
__device__ __forceinline__ lds_v4i* to_lds(void* p) {
    // generic LDS pointer: addr[31:0] is the LDS offset (aperture in high bits)
    return (lds_v4i*)(unsigned int)(uintptr_t)p;
}

__device__ __forceinline__ void wait_async0() {
#if __has_builtin(__builtin_amdgcn_s_wait_asynccnt)
    __builtin_amdgcn_s_wait_asynccnt(0);
#else
    asm volatile("s_wait_asynccnt 0x0" ::: "memory");
#endif
}

// ---------------------------------------------------------------------------
__global__ __launch_bounds__(256) void k_normalize(const float* __restrict__ x,
                                                   __bf16* __restrict__ xb,
                                                   float* __restrict__ invn) {
    const int row = blockIdx.x;
    const int tid = threadIdx.x;
    const float* xr = x + (size_t)row * DIM;
    float vals[4];
    float s = 0.f;
#pragma unroll
    for (int t = 0; t < 4; ++t) {
        float v = xr[tid + 256 * t];
        vals[t] = v;
        s += v * v;
    }
#pragma unroll
    for (int off = 16; off; off >>= 1) s += __shfl_xor(s, off);
    __shared__ float red[8];
    __shared__ float scale_sh;
    if ((tid & 31) == 0) red[tid >> 5] = s;
    __syncthreads();
    if (tid == 0) {
        float tot = 0.f;
#pragma unroll
        for (int w = 0; w < 8; ++w) tot += red[w];
        float sc = 1.0f / fmaxf(sqrtf(tot), EPSF);
        scale_sh = sc;
        invn[row] = sc;
    }
    __syncthreads();
    const float sc = scale_sh;
    __bf16* xbr = xb + (size_t)row * DIM;
#pragma unroll
    for (int t = 0; t < 4; ++t) xbr[tid + 256 * t] = (__bf16)(vals[t] * sc);
}

// ---------------------------------------------------------------------------
__global__ __launch_bounds__(256) void k_init_keys(unsigned long long* __restrict__ keys) {
    int i = blockIdx.x * 256 + threadIdx.x;
    if (i < N_ROWS) keys[i] = 0ull;   // below any dot >= -1 after monotonic mapping
}

// monotonic float->uint map: order-preserving for atomicMax
__device__ __forceinline__ unsigned mono_bits(float v) {
    unsigned u = __float_as_uint(v);
    return (u & 0x80000000u) ? ~u : (u | 0x80000000u);
}

// ---------------------------------------------------------------------------
__global__ __launch_bounds__(256) void k_argmax_gram(const __bf16* __restrict__ xb,
                                                     unsigned long long* __restrict__ keys) {
    extern __shared__ char smem[];
    char* As  = smem;                                          // 64 rows x 2064 B
    char* Bs0 = smem + ROWS_PER_BLOCK * A_STRIDE_B;            // 128 cols x 80 B
    char* Bs1 = Bs0 + COLS_PER_GROUP * B_COL_STRIDE_B;

    const int tid  = threadIdx.x;
    const int lane = tid & 31;
    const int wave = tid >> 5;
    const int m    = lane & 15;   // N (B/C) or M (A) position within 16
    const int h    = lane >> 4;   // half-group select
    const int rw   = wave & 3;    // row sub-strip (16 rows each)
    const int ch   = wave >> 2;   // column half (64 cols each) of the group

    const int row_base   = blockIdx.x * ROWS_PER_BLOCK;
    const int chunk_base = blockIdx.y * COL_CHUNK;

    // this thread's two B-panel chunks (fixed col/sub per thread)
    const int bcol0 = tid >> 2;             // chunk 0: col = tid/4
    const int bsub0 = tid & 3;
    const int bcol1 = (tid + 256) >> 2;     // chunk 1
    const int bsub1 = (tid + 256) & 3;

    // ---- stage the whole 64-row A strip (bf16) into LDS once ----
    {
        const uint4* gsrc = (const uint4*)(xb + (size_t)row_base * DIM);
#pragma unroll
        for (int c = tid; c < ROWS_PER_BLOCK * (DIM / 8); c += 256) {
            int r  = c >> 7;              // DIM/8 = 128 16B-chunks per row
            int kc = c & 127;
            uint4 d = gsrc[r * (DIM / 8) + kc];
            *(uint4*)(As + r * A_STRIDE_B + kc * 16) = d;
        }
    }
    __syncthreads();

    // running per-lane argmax: VGPR v of C holds row M = v + 8*h, col N = m
    float bestv[8];
    int   bestj[8];
#pragma unroll
    for (int v = 0; v < 8; ++v) { bestv[v] = -2.0f; bestj[v] = 0; }

#if USE_ASYNC_LDS
    // issue async staging of the first panel (g=0, kk=0) into Bs0
    {
        const int col0 = chunk_base;
        __builtin_amdgcn_global_load_async_to_lds_b128(
            to_glb(xb + (size_t)(col0 + bcol0) * DIM + 0 + bsub0 * 8),
            to_lds(Bs0 + bcol0 * B_COL_STRIDE_B + bsub0 * 16), 0, 0);
        __builtin_amdgcn_global_load_async_to_lds_b128(
            to_glb(xb + (size_t)(col0 + bcol1) * DIM + 0 + bsub1 * 8),
            to_lds(Bs0 + bcol1 * B_COL_STRIDE_B + bsub1 * 16), 0, 0);
    }
#else
    // register prefetch of the first panel
    uint4 pre0, pre1;
    {
        const int col0 = chunk_base;
        pre0 = *(const uint4*)(xb + (size_t)(col0 + bcol0) * DIM + 0 + bsub0 * 8);
        pre1 = *(const uint4*)(xb + (size_t)(col0 + bcol1) * DIM + 0 + bsub1 * 8);
    }
#endif

    for (int g = 0; g < N_GROUPS; ++g) {
        const int col0 = chunk_base + g * COLS_PER_GROUP;

        v8f acc[4];
#pragma unroll
        for (int t = 0; t < 4; ++t) acc[t] = (v8f){0.f,0.f,0.f,0.f,0.f,0.f,0.f,0.f};

        for (int kk = 0; kk < DIM; kk += KSTEP) {
            char* Bs  = ((kk >> 5) & 1) ? Bs1 : Bs0;   // current panel
            char* Bsn = ((kk >> 5) & 1) ? Bs0 : Bs1;   // next panel buffer
            const bool last  = (g == N_GROUPS - 1) && (kk == DIM - KSTEP);
            const int  nkk   = (kk + KSTEP < DIM) ? kk + KSTEP : 0;
            const int  ncol0 = (kk + KSTEP < DIM) ? col0 : col0 + COLS_PER_GROUP;

#if USE_ASYNC_LDS
            // current panel was issued one step ago; wait + sync, then issue next
            wait_async0();
            __syncthreads();
            if (!last) {
                __builtin_amdgcn_global_load_async_to_lds_b128(
                    to_glb(xb + (size_t)(ncol0 + bcol0) * DIM + nkk + bsub0 * 8),
                    to_lds(Bsn + bcol0 * B_COL_STRIDE_B + bsub0 * 16), 0, 0);
                __builtin_amdgcn_global_load_async_to_lds_b128(
                    to_glb(xb + (size_t)(ncol0 + bcol1) * DIM + nkk + bsub1 * 8),
                    to_lds(Bsn + bcol1 * B_COL_STRIDE_B + bsub1 * 16), 0, 0);
            }
#else
            // store prefetched panel, then immediately issue loads for the next one
            *(uint4*)(Bs + bcol0 * B_COL_STRIDE_B + bsub0 * 16) = pre0;
            *(uint4*)(Bs + bcol1 * B_COL_STRIDE_B + bsub1 * 16) = pre1;
            if (!last) {
                pre0 = *(const uint4*)(xb + (size_t)(ncol0 + bcol0) * DIM + nkk + bsub0 * 8);
                pre1 = *(const uint4*)(xb + (size_t)(ncol0 + bcol1) * DIM + nkk + bsub1 * 8);
            }
#if __has_builtin(__builtin_amdgcn_sched_barrier)
            __builtin_amdgcn_sched_barrier(0);   // keep prefetch issued before compute
#endif
            __syncthreads();
#endif

            // A fragment (16x32 bf16 layout): lane(m,h) -> row m, K = kk+8h..+7, kk+16+8h..+7
            U32x8 au;
            {
                const char* ap = As + (rw * 16 + m) * A_STRIDE_B + kk * 2 + h * 16;
                au.lo = *(const uint4*)(ap);
                au.hi = *(const uint4*)(ap + 32);
            }
            v16bf a = __builtin_bit_cast(v16bf, au);

            // load all four B fragments first (single DS-wait), then 4 WMMAs
            v16bf b[4];
#pragma unroll
            for (int t = 0; t < 4; ++t) {
                const char* bp = Bs + (ch * 64 + t * 16 + m) * B_COL_STRIDE_B + h * 16;
                U32x8 bu;
                bu.lo = *(const uint4*)(bp);
                bu.hi = *(const uint4*)(bp + 32);
                b[t] = __builtin_bit_cast(v16bf, bu);
            }
#pragma unroll
            for (int t = 0; t < 4; ++t) {
                acc[t] = __builtin_amdgcn_wmma_f32_16x16x32_bf16(
                    /*neg_a=*/false, a, /*neg_b=*/false, b[t],
                    /*c_mod=*/(short)0, acc[t], /*reuse_a=*/false, /*reuse_b=*/false);
            }
        }

        // fold this 16x64 (per wave) C block into the running argmax, skip diagonal
#pragma unroll
        for (int t = 0; t < 4; ++t) {
            const int j = col0 + ch * 64 + t * 16 + m;
#pragma unroll
            for (int v = 0; v < 8; ++v) {
                const int i = row_base + rw * 16 + v + 8 * h;
                float val = acc[t][v];
                bool upd = (j != i) &&
                           ((val > bestv[v]) || (val == bestv[v] && j < bestj[v]));
                if (upd) { bestv[v] = val; bestj[v] = j; }
            }
        }
    }

    // reduce across the 16 lanes sharing each C row, then merge globally
#pragma unroll
    for (int v = 0; v < 8; ++v) {
        float bv = bestv[v];
        int   bj = bestj[v];
#pragma unroll
        for (int off = 1; off < 16; off <<= 1) {
            float ov = __shfl_xor(bv, off);
            int   oj = __shfl_xor(bj, off);
            if (ov > bv || (ov == bv && oj < bj)) { bv = ov; bj = oj; }
        }
        if (m == 0) {
            const int i = row_base + rw * 16 + v + 8 * h;
            unsigned long long key =
                ((unsigned long long)mono_bits(bv) << 32) | (unsigned)bj;
            atomicMax(&keys[i], key);
        }
    }
}

// ---------------------------------------------------------------------------
__global__ __launch_bounds__(256) void k_distance(const float* __restrict__ x,
                                                  const float* __restrict__ invn,
                                                  const unsigned long long* __restrict__ keys,
                                                  float* __restrict__ rowloss) {
    const int i   = blockIdx.x;
    const int tid = threadIdx.x;
    const int j   = (int)(unsigned)(keys[i] & 0xffffffffull);
    const float si = invn[i];
    const float sj = invn[j];
    const float* xi = x + (size_t)i * DIM;
    const float* xj = x + (size_t)j * DIM;
    float s = 0.f;
#pragma unroll
    for (int t = 0; t < 4; ++t) {
        int k = tid + 256 * t;
        float d = xi[k] * si - xj[k] * sj + EPSF;
        s += d * d;
    }
#pragma unroll
    for (int off = 16; off; off >>= 1) s += __shfl_xor(s, off);
    __shared__ float red[8];
    if ((tid & 31) == 0) red[tid >> 5] = s;
    __syncthreads();
    if (tid == 0) {
        float tot = 0.f;
#pragma unroll
        for (int w = 0; w < 8; ++w) tot += red[w];
        rowloss[i] = -logf(sqrtf(tot) + EPSF);
    }
}

// ---------------------------------------------------------------------------
__global__ __launch_bounds__(256) void k_mean(const float* __restrict__ rowloss,
                                              float* __restrict__ out) {
    const int tid = threadIdx.x;
    float s = 0.f;
    for (int i = tid; i < N_ROWS; i += 256) s += rowloss[i];
#pragma unroll
    for (int off = 16; off; off >>= 1) s += __shfl_xor(s, off);
    __shared__ float red[8];
    if ((tid & 31) == 0) red[tid >> 5] = s;
    __syncthreads();
    if (tid == 0) {
        float tot = 0.f;
#pragma unroll
        for (int w = 0; w < 8; ++w) tot += red[w];
        out[0] = tot / (float)N_ROWS;
    }
}

// ---------------------------------------------------------------------------
extern "C" void kernel_launch(void* const* d_in, const int* in_sizes, int n_in,
                              void* d_out, int out_size, void* d_ws, size_t ws_size,
                              hipStream_t stream) {
    const float* x = (const float*)d_in[0];
    char* ws = (char*)d_ws;

    __bf16*             xb      = (__bf16*)ws;
    float*              invn    = (float*)(ws + (size_t)N_ROWS * DIM * 2);
    unsigned long long* keys    = (unsigned long long*)(ws + (size_t)N_ROWS * DIM * 2
                                                           + (size_t)N_ROWS * 4);
    float*              rowloss = (float*)(ws + (size_t)N_ROWS * DIM * 2
                                              + (size_t)N_ROWS * 4
                                              + (size_t)N_ROWS * 8);
    float* out = (float*)d_out;

    k_normalize<<<N_ROWS, 256, 0, stream>>>(x, xb, invn);
    k_init_keys<<<N_ROWS / 256, 256, 0, stream>>>(keys);

    const size_t lds_bytes = (size_t)ROWS_PER_BLOCK * A_STRIDE_B
                           + 2ull * COLS_PER_GROUP * B_COL_STRIDE_B;   // 152576 B
    dim3 grid(N_ROWS / ROWS_PER_BLOCK, N_ROWS / COL_CHUNK);            // 128 x 8
    k_argmax_gram<<<grid, 256, lds_bytes, stream>>>(xb, keys);

    k_distance<<<N_ROWS, 256, 0, stream>>>(x, invn, keys, rowloss);
    k_mean<<<1, 256, 0, stream>>>(rowloss, out);
}
